// XNet_68813966016551
// MI455X (gfx1250) — compile-verified, matrix operand order checked by reading, not run on hardware
//
#include <hip/hip_runtime.h>
#include <hip/hip_bf16.h>

// ---------------- constants ----------------
namespace {
constexpr int B  = 2;
constexpr int C  = 64;
constexpr int HH = 48;
constexpr int N  = HH * HH;      // 2304
constexpr int N1 = N + 1;        // 2305
constexpr float SCALE = 0.125f;  // 1/sqrt(64)
constexpr float INV_REG = 10.0f; // 1/0.1
constexpr float W_IN  = 0.8f / (float)N;
constexpr float W_PAD = 0.2f;
// quantile: pos = 0.75*(N*N-1) = 3981311.25
constexpr unsigned QK0 = 3981311u;
constexpr unsigned QK1 = 3981312u;
constexpr int NFRAG = 288;            // fragments per packed matrix (144*2 or 72*4)
constexpr int FRAGSZ = 512;           // halves per fragment (32 lanes x 16)
}

typedef __attribute__((ext_vector_type(16))) _Float16 v16h;
typedef __attribute__((ext_vector_type(8)))  float    v8f;

// ---------------- WMMA helpers ----------------
__device__ __forceinline__ v8f wmma32(v16h a, v16h b, v8f c) {
  return __builtin_amdgcn_wmma_f32_16x16x32_f16(false, a, false, b, (short)0, c,
                                                false, false);
}

// A/B fragment where element[x][k] lives at base[k*ld + x] (k-major rows).
// A-matrix 16x32 layout: lanes 0-15 rows m, halves 0..7 -> K 0..7, 8..15 -> K 16..23;
// lanes 16-31 same rows, K 8..15 / 24..31. B mirrors with x = column.
__device__ __forceinline__ v16h frag_k(const float* __restrict__ base, int ld,
                                       int x0, int k0, int lane) {
  const int x  = x0 + (lane & 15);
  const int kb = (lane < 16) ? 0 : 8;
  v16h f;
#pragma unroll
  for (int h = 0; h < 16; ++h) {
    const int k = k0 + kb + ((h < 8) ? h : (h + 8));
    f[h] = (_Float16)base[(size_t)k * ld + x];
  }
  return f;
}

// Fragment where element[x][k] lives at base[x*ld + k] (x-major rows).
__device__ __forceinline__ v16h frag_x(const float* __restrict__ base, int ld,
                                       int x0, int k0, int lane) {
  const int x  = x0 + (lane & 15);
  const int kb = (lane < 16) ? 0 : 8;
  v16h f;
#pragma unroll
  for (int h = 0; h < 16; ++h) {
    const int k = k0 + kb + ((h < 8) ? h : (h + 8));
    f[h] = (_Float16)base[(size_t)x * ld + k];
  }
  return f;
}

// Load a pre-packed fragment: 32 contiguous bytes per lane -> 2x global_load_b128.
__device__ __forceinline__ v16h ldfrag(const _Float16* __restrict__ p, int f,
                                       int lane) {
  return *(const v16h*)(p + ((size_t)f << 9) + (lane << 4));
}

__device__ __forceinline__ float rmax16(float x) {
#pragma unroll
  for (int m = 1; m < 16; m <<= 1) x = fmaxf(x, __shfl_xor(x, m));
  return x;
}
__device__ __forceinline__ float rsum16(float x) {
#pragma unroll
  for (int m = 1; m < 16; m <<= 1) x += __shfl_xor(x, m);
  return x;
}

// ---------------- per-token squared norms ----------------
__global__ void sqnorm_kernel(const float* __restrict__ inp_a,
                              const float* __restrict__ inp_b,
                              float* __restrict__ sqA, float* __restrict__ sqB) {
  int t = blockIdx.x * blockDim.x + threadIdx.x;
  if (t >= 2 * B * N) return;
  int n = t % N;
  int b = (t / N) % B;
  int which = t / (B * N);
  const float* x = (which ? inp_b : inp_a) + (size_t)b * C * N;
  float s = 0.f;
#pragma unroll 8
  for (int c = 0; c < C; ++c) {
    float v = x[(size_t)c * N + n];
    s += v * v;
  }
  (which ? sqB : sqA)[b * N + n] = s;
}

// ---------------- fragment pre-packing (one-time f32 gather -> f16 frags) ----
// pack inputs for the cost GEMM: role = which-tensor (0=a,1=b), frag_k over tokens
__global__ __launch_bounds__(32) void pack_inp_kernel(
    const float* __restrict__ inp_a, const float* __restrict__ inp_b,
    _Float16* __restrict__ pk) {
  const int f = blockIdx.x;            // 0..287: (token-tile)*2 + chunk
  const int y = blockIdx.y;            // which*2 + b
  const int which = y >> 1, b = y & 1;
  const int lane = threadIdx.x;
  const float* src = (which ? inp_b : inp_a) + (size_t)b * C * N;
  const int xt = f >> 1, k0 = (f & 1) * 32;
  v16h v = frag_k(src, N, xt * 16, k0, lane);
  *(v16h*)(pk + (((size_t)which * B + b) * NFRAG + f) * FRAGSZ + lane * 16) = v;
}

// pack q/k/v for the attention GEMMs.
// roles: 0:Q(dir0)=qB  1:K(dir0)=kA  2:V(dir0)=vA  3:Q(dir1)=qA  4:K(dir1)=kB  5:V(dir1)=vB
// Q/K use frag_k layout (f = token-tile*2 + chunk); V uses frag_x (f = keyblock*4 + cg).
__global__ __launch_bounds__(32) void pack_qkv_kernel(
    const float* __restrict__ qkvA, const float* __restrict__ qkvB,
    _Float16* __restrict__ pk) {
  const int f = blockIdx.x;            // 0..287
  const int y = blockIdx.y;            // r*2 + b
  const int b = y & 1, r = y >> 1;
  const int lane = threadIdx.x;
  const float* src;
  int type; // 0 = frag_k, 1 = frag_x
  switch (r) {
    case 0: src = qkvB + ((size_t)b * 3 * C + 0) * N;     type = 0; break;
    case 1: src = qkvA + ((size_t)b * 3 * C + C) * N;     type = 0; break;
    case 2: src = qkvA + ((size_t)b * 3 * C + 2 * C) * N; type = 1; break;
    case 3: src = qkvA + ((size_t)b * 3 * C + 0) * N;     type = 0; break;
    case 4: src = qkvB + ((size_t)b * 3 * C + C) * N;     type = 0; break;
    default: src = qkvB + ((size_t)b * 3 * C + 2 * C) * N; type = 1; break;
  }
  v16h v;
  if (type == 0) {
    const int xt = f >> 1, k0 = (f & 1) * 32;
    v = frag_k(src, N, xt * 16, k0, lane);
  } else {
    const int kb = f >> 2, cg = f & 3;
    v = frag_x(src + (size_t)cg * 16 * N, N, 0, kb * 32, lane);
  }
  *(v16h*)(pk + (((size_t)r * B + b) * NFRAG + f) * FRAGSZ + lane * 16) = v;
}

// ---------------- cost matrix via WMMA: Kbuf interior <- max(||s||^2+||t||^2-2 s.t, 0)
__global__ __launch_bounds__(32) void cost_kernel(
    const _Float16* __restrict__ pkInp, const float* __restrict__ sqA,
    const float* __restrict__ sqB, float* __restrict__ Ka,
    float* __restrict__ Kb) {
  const int it = blockIdx.x, jt = blockIdx.y;
  const int z = blockIdx.z, dir = z >> 1, b = z & 1;
  const float *sqs, *sqt;
  const _Float16 *pa, *pb;
  float* Kd;
  const _Float16* pkA = pkInp + ((size_t)0 * B + b) * NFRAG * FRAGSZ;
  const _Float16* pkB = pkInp + ((size_t)1 * B + b) * NFRAG * FRAGSZ;
  if (dir == 0) {
    pa = pkA; pb = pkB; sqs = sqA + b * N; sqt = sqB + b * N;
    Kd = Ka + (size_t)b * N1 * N1;
  } else {
    pa = pkB; pb = pkA; sqs = sqB + b * N; sqt = sqA + b * N;
    Kd = Kb + (size_t)b * N1 * N1;
  }
  const int lane = threadIdx.x;
  const int i0 = it * 16, j0 = jt * 16;
  v8f acc = {};
  acc = wmma32(ldfrag(pa, it * 2 + 0, lane), ldfrag(pb, jt * 2 + 0, lane), acc);
  acc = wmma32(ldfrag(pa, it * 2 + 1, lane), ldfrag(pb, jt * 2 + 1, lane), acc);
  const int col = lane & 15, hi = lane >> 4;
  const float st = sqt[j0 + col];
#pragma unroll
  for (int g = 0; g < 8; ++g) {
    const int row = g + 8 * hi;
    float cst = fmaxf(sqs[i0 + row] + st - 2.0f * acc[g], 0.0f);
    Kd[(size_t)(i0 + row) * N1 + (j0 + col)] = cst;
  }
}

// ---------------- quantile: 4-pass radix select --------------
__global__ void qinit_kernel(unsigned* __restrict__ qstate) {
  int s = threadIdx.x;
  if (s < 4) {
    qstate[s * 4 + 0] = 0u;
    qstate[s * 4 + 1] = (s & 1) ? QK1 : QK0;
    qstate[s * 4 + 2] = 0u;
  }
}
__global__ void qzero_kernel(unsigned* __restrict__ hist) {
  hist[threadIdx.x] = 0u;
}
__global__ void qhist_kernel(const float* __restrict__ Km,
                             const unsigned* __restrict__ qstate,
                             unsigned* __restrict__ hist, unsigned maskHigh,
                             unsigned shift, int slot) {
  __shared__ unsigned h[256];
  h[threadIdx.x] = 0u;
  __syncthreads();
  const unsigned pref = qstate[slot * 4 + 0];
  size_t idx = (size_t)blockIdx.x * 256 + threadIdx.x;
  if (idx < (size_t)N * N) {
    size_t i = idx / N, j = idx - i * N;
    unsigned bits = __float_as_uint(Km[i * (size_t)N1 + j]); // cost >= 0: monotone
    if ((bits & maskHigh) == pref) atomicAdd(&h[(bits >> shift) & 255u], 1u);
  }
  __syncthreads();
  if (h[threadIdx.x]) atomicAdd(&hist[threadIdx.x], h[threadIdx.x]);
}
__global__ void qscan_kernel(unsigned* __restrict__ qstate,
                             const unsigned* __restrict__ hist, unsigned shift,
                             int slot) {
  unsigned k = qstate[slot * 4 + 1];
  unsigned pref = qstate[slot * 4 + 0];
  unsigned cum = 0;
  int chosen = 255;
  for (int bq = 0; bq < 256; ++bq) {
    unsigned hv = hist[bq];
    if (cum + hv > k) { chosen = bq; break; }
    cum += hv;
  }
  qstate[slot * 4 + 1] = k - cum;
  pref |= ((unsigned)chosen) << shift;
  qstate[slot * 4 + 0] = pref;
  if (shift == 0) qstate[slot * 4 + 2] = pref; // exact value bits
}

// ---------------- borders + K transform ----------------
__global__ void border_kernel(float* __restrict__ Ka, float* __restrict__ Kb,
                              const unsigned* __restrict__ qstate) {
  int t = blockIdx.x * blockDim.x + threadIdx.x;
  if (t >= 4 * N1) return;
  int i = t % N1, z = t / N1, dir = z >> 1, b = z & 1;
  float r0 = __uint_as_float(qstate[(dir * 2 + 0) * 4 + 2]);
  float r1 = __uint_as_float(qstate[(dir * 2 + 1) * 4 + 2]);
  float vc = r0 + 0.25f * (r1 - r0); // linear interp at 0.75*(M-1)
  float* Km = (dir ? Kb : Ka) + (size_t)b * N1 * N1;
  float val = (i < N) ? vc : 0.0f;
  Km[(size_t)i * N1 + N] = val;
  Km[(size_t)N * N1 + i] = val;
}
__global__ void ktransform_kernel(float* __restrict__ Kall) {
  size_t idx = (size_t)blockIdx.x * 256 + threadIdx.x;
  if (idx >= (size_t)2 * B * N1 * N1) return;
  Kall[idx] = fmaxf(__expf(-Kall[idx] * INV_REG), 1e-8f);
}

// ---------------- Sinkhorn ----------------
__global__ void initv_kernel(float* __restrict__ v) {
  int t = blockIdx.x * blockDim.x + threadIdx.x;
  if (t < 4 * N1) v[t] = 1.0f;
}
// u[row] = w(row) / (sum_c K[row][c] * v[c] + eps)   (coalesced row read)
__global__ void ustep_kernel(const float* __restrict__ Ka,
                             const float* __restrict__ Kb,
                             const float* __restrict__ v, float* __restrict__ u) {
  const int row = blockIdx.x;
  const int p = blockIdx.y;
  const float* Km = (p < 2 ? Ka : Kb) + (size_t)(p & 1) * N1 * N1;
  const float* vp = v + p * N1;
  float s = 0.f;
  for (int c = threadIdx.x; c < N1; c += 256)
    s += Km[(size_t)row * N1 + c] * vp[c];
  __shared__ float red[256];
  red[threadIdx.x] = s;
  __syncthreads();
  for (int st = 128; st; st >>= 1) {
    if (threadIdx.x < st) red[threadIdx.x] += red[threadIdx.x + st];
    __syncthreads();
  }
  if (threadIdx.x == 0) {
    float w = (row < N) ? W_IN : W_PAD;
    u[p * N1 + row] = w / (red[0] + 1e-6f);
  }
}
// v[col] = w(col) / (sum_r K[r][col] * u[r] + eps)  (64 cols x 4 row-strides, coalesced)
__global__ void vstep_kernel(const float* __restrict__ Ka,
                             const float* __restrict__ Kb,
                             const float* __restrict__ u, float* __restrict__ v) {
  const int p = blockIdx.y;
  const int cx = threadIdx.x & 63, ry = threadIdx.x >> 6;
  const int col = blockIdx.x * 64 + cx;
  const float* Km = (p < 2 ? Ka : Kb) + (size_t)(p & 1) * N1 * N1;
  const float* up = u + p * N1;
  float s = 0.f;
  if (col < N1)
    for (int r = ry; r < N1; r += 4) s += Km[(size_t)r * N1 + col] * up[r];
  __shared__ float red[256];
  red[threadIdx.x] = s;
  __syncthreads();
  if (ry == 0 && col < N1) {
    float tot = red[cx] + red[cx + 64] + red[cx + 128] + red[cx + 192];
    float w = (col < N) ? W_IN : W_PAD;
    v[p * N1 + col] = w / (tot + 1e-6f);
  }
}
// invden[j] = 1 / sum_{i<N} u[i]*K[i][j]   (v cancels in the normalized plan)
__global__ void den_kernel(const float* __restrict__ Ka,
                           const float* __restrict__ Kb,
                           const float* __restrict__ u,
                           float* __restrict__ invden) {
  const int p = blockIdx.y;
  const int cx = threadIdx.x & 63, ry = threadIdx.x >> 6;
  const int col = blockIdx.x * 64 + cx; // < N (grid 36*64)
  const float* Km = (p < 2 ? Ka : Kb) + (size_t)(p & 1) * N1 * N1;
  const float* up = u + p * N1;
  float s = 0.f;
  for (int r = ry; r < N; r += 4) s += Km[(size_t)r * N1 + col] * up[r];
  __shared__ float red[256];
  red[threadIdx.x] = s;
  __syncthreads();
  if (ry == 0) {
    float tot = red[cx] + red[cx + 64] + red[cx + 128] + red[cx + 192];
    invden[p * N + col] = 1.0f / (tot + 1e-30f);
  }
}

// ---------------- convolutions ----------------
__global__ void conv1x1_qkv_kernel(const float* __restrict__ x,
                                   const float* __restrict__ w,
                                   float* __restrict__ out) {
  int t = blockIdx.x * blockDim.x + threadIdx.x;
  if (t >= B * 3 * C * N) return;
  int n = t % N, oc = (t / N) % (3 * C), b = t / (N * 3 * C);
  const float* xb = x + (size_t)b * C * N;
  const float* wr = w + (size_t)oc * C;
  float s = 0.f;
#pragma unroll 8
  for (int ic = 0; ic < C; ++ic) s += wr[ic] * xb[(size_t)ic * N + n];
  out[t] = s;
}
__global__ void dwconv_kernel(const float* __restrict__ x,
                              const float* __restrict__ w,
                              float* __restrict__ out) {
  int t = blockIdx.x * blockDim.x + threadIdx.x;
  if (t >= B * 3 * C * N) return;
  int wx = t % HH, hy = (t / HH) % HH, c = (t / N) % (3 * C), b = t / (N * 3 * C);
  const float* xc = x + ((size_t)b * 3 * C + c) * N;
  const float* wc = w + (size_t)c * 9;
  float s = 0.f;
#pragma unroll
  for (int dy = 0; dy < 3; ++dy) {
    int yy = hy + dy - 1;
    if ((unsigned)yy >= (unsigned)HH) continue;
#pragma unroll
    for (int dx = 0; dx < 3; ++dx) {
      int xx = wx + dx - 1;
      if ((unsigned)xx >= (unsigned)HH) continue;
      s += wc[dy * 3 + dx] * xc[yy * HH + xx];
    }
  }
  out[t] = s;
}
__global__ void outconv_kernel(const float* __restrict__ o,
                               const float* __restrict__ w,
                               const float* __restrict__ bias,
                               const float* __restrict__ resid,
                               float* __restrict__ dst) {
  int t = blockIdx.x * blockDim.x + threadIdx.x;
  if (t >= B * C * N) return;
  int n = t % N, c = (t / N) % C, b = t / (N * C);
  const float* ob = o + (size_t)b * C * N;
  const float* wr = w + (size_t)c * C;
  float s = 0.f;
#pragma unroll 8
  for (int ic = 0; ic < C; ++ic) s += wr[ic] * ob[(size_t)ic * N + n];
  dst[t] = s + bias[c] + resid[t];
}

// ---------------- fused flash attention + OT bias (WMMA) ----------------
// One wave per 16-query tile. S = scale*Q.K^T + coef*u[i]*K[i,j]*invden[j],
// online softmax, P.V via WMMA, never materializing the NxN attention matrix.
// Q/K/V come from pre-packed f16 fragment buffers (2x b128 per fragment).
__global__ __launch_bounds__(32) void attn_kernel(
    const _Float16* __restrict__ pkQkv, const float* __restrict__ Ka,
    const float* __restrict__ Kb, const float* __restrict__ u,
    const float* __restrict__ invden, const float* __restrict__ alpha_p,
    const float* __restrict__ beta_p, float* __restrict__ oA,
    float* __restrict__ oB) {
  const int jt = blockIdx.x;
  const int z = blockIdx.y, dir = z >> 1, b = z & 1;
  const size_t MAT = (size_t)NFRAG * FRAGSZ;
  const _Float16* Qp = pkQkv + ((size_t)(dir * 3 + 0) * B + b) * MAT;
  const _Float16* Kp = pkQkv + ((size_t)(dir * 3 + 1) * B + b) * MAT;
  const _Float16* Vp = pkQkv + ((size_t)(dir * 3 + 2) * B + b) * MAT;
  const float* Km = (dir ? Kb : Ka) + (size_t)b * N1 * N1;
  const float* up = u + (dir * 2 + b) * N1;
  const float* dnp = invden + (dir * 2 + b) * N;
  const float coef = dir ? beta_p[0] : alpha_p[0];

  const int lane = threadIdx.x;
  const int col = lane & 15, hi = lane >> 4;
  const int j0 = jt * 16;
  __shared__ float lds[16 * 32];

  v16h qa0 = ldfrag(Qp, jt * 2 + 0, lane);
  v16h qa1 = ldfrag(Qp, jt * 2 + 1, lane);

  float dinv[8];
#pragma unroll
  for (int g = 0; g < 8; ++g) dinv[g] = dnp[j0 + g + 8 * hi] * coef;

  float mrun[8], lrun[8];
  v8f acc0 = {}, acc1 = {}, acc2 = {}, acc3 = {};
#pragma unroll
  for (int g = 0; g < 8; ++g) { mrun[g] = -1e30f; lrun[g] = 0.f; }

  for (int t = 0; t < N / 32; ++t) {
    const int i0 = t * 32;
    // ---- S = Q.K^T (two 16-key subtiles, C=64 chained over k=32 chunks)
    v8f s0 = {}, s1 = {};
    s0 = wmma32(qa0, ldfrag(Kp, (2 * t + 0) * 2 + 0, lane), s0);
    s0 = wmma32(qa1, ldfrag(Kp, (2 * t + 0) * 2 + 1, lane), s0);
    s1 = wmma32(qa0, ldfrag(Kp, (2 * t + 1) * 2 + 0, lane), s1);
    s1 = wmma32(qa1, ldfrag(Kp, (2 * t + 1) * 2 + 1, lane), s1);
    // ---- add OT transport bias: coef * u[i]*K[i,j]/den[j]
    const float u0 = up[i0 + col], u1 = up[i0 + 16 + col];
#pragma unroll
    for (int g = 0; g < 8; ++g) {
      const int j = j0 + g + 8 * hi;
      s0[g] = s0[g] * SCALE + u0 * Km[(size_t)(i0 + col) * N1 + j] * dinv[g];
      s1[g] = s1[g] * SCALE + u1 * Km[(size_t)(i0 + 16 + col) * N1 + j] * dinv[g];
    }
    // ---- online softmax
    float f[8];
#pragma unroll
    for (int g = 0; g < 8; ++g) {
      float rm = rmax16(fmaxf(s0[g], s1[g]));
      float mn = fmaxf(mrun[g], rm);
      f[g] = __expf(mrun[g] - mn);
      mrun[g] = mn;
      float p0 = __expf(s0[g] - mn), p1 = __expf(s1[g] - mn);
      lrun[g] = lrun[g] * f[g] + rsum16(p0 + p1);
      lds[(g + 8 * hi) * 32 + col] = p0;
      lds[(g + 8 * hi) * 32 + 16 + col] = p1;
    }
#pragma unroll
    for (int g = 0; g < 8; ++g) {
      acc0[g] *= f[g]; acc1[g] *= f[g]; acc2[g] *= f[g]; acc3[g] *= f[g];
    }
    __syncthreads();
    // ---- P.V : P (16x32) from LDS as A-frag; V subtiles (32 keys x 16 ch) packed
    v16h pa = frag_x(lds, 32, 0, 0, lane);
    acc0 = wmma32(pa, ldfrag(Vp, t * 4 + 0, lane), acc0);
    acc1 = wmma32(pa, ldfrag(Vp, t * 4 + 1, lane), acc1);
    acc2 = wmma32(pa, ldfrag(Vp, t * 4 + 2, lane), acc2);
    acc3 = wmma32(pa, ldfrag(Vp, t * 4 + 3, lane), acc3);
    __syncthreads();
  }
  float* o = (dir ? oB : oA) + (size_t)b * C * N;
#pragma unroll
  for (int g = 0; g < 8; ++g) {
    const int m = g + 8 * hi;
    const float il = 1.0f / lrun[g];
    o[(size_t)(0 * 16 + col) * N + j0 + m] = acc0[g] * il;
    o[(size_t)(1 * 16 + col) * N + j0 + m] = acc1[g] * il;
    o[(size_t)(2 * 16 + col) * N + j0 + m] = acc2[g] * il;
    o[(size_t)(3 * 16 + col) * N + j0 + m] = acc3[g] * il;
  }
}

// ---------------- host-side orchestration ----------------
extern "C" void kernel_launch(void* const* d_in, const int* in_sizes, int n_in,
                              void* d_out, int out_size, void* d_ws,
                              size_t ws_size, hipStream_t stream) {
  (void)in_sizes; (void)n_in; (void)out_size; (void)ws_size;
  const float* inp_a   = (const float*)d_in[0];
  const float* inp_b   = (const float*)d_in[1];
  const float* alpha_p = (const float*)d_in[2];
  const float* beta_p  = (const float*)d_in[3];
  const float* qkvA_w1 = (const float*)d_in[4];
  const float* qkvA_dw = (const float*)d_in[5];
  const float* outA_w  = (const float*)d_in[6];
  const float* outA_b  = (const float*)d_in[7];
  const float* qkvB_w1 = (const float*)d_in[8];
  const float* qkvB_dw = (const float*)d_in[9];
  const float* outB_w  = (const float*)d_in[10];
  const float* outB_b  = (const float*)d_in[11];
  float* out = (float*)d_out;

  // workspace layout (floats); K buffers are L2-resident (85 MB < 192 MB L2)
  const size_t SK = (size_t)B * N1 * N1;
  float* ws = (float*)d_ws;
  float* Ka = ws;
  float* Kb = Ka + SK;
  float* u = Kb + SK;                 // 4*N1
  float* v = u + 4 * N1;              // 4*N1
  float* invden = v + 4 * N1;         // 4*N
  float* sqA = invden + 4 * N;        // B*N
  float* sqB = sqA + B * N;           // B*N
  float* tmp1A = sqB + B * N;         // B*3C*N
  float* tmp1B = tmp1A + (size_t)B * 3 * C * N;
  float* qkvA = tmp1B + (size_t)B * 3 * C * N;
  float* qkvB = qkvA + (size_t)B * 3 * C * N;
  float* oA = qkvB + (size_t)B * 3 * C * N; // B*C*N
  float* oB = oA + (size_t)B * C * N;
  unsigned* qstate = (unsigned*)(oB + (size_t)B * C * N); // 16 uints
  unsigned* hist = qstate + 16;                           // 256 uints
  uintptr_t pb = ((uintptr_t)(hist + 256) + 31) & ~(uintptr_t)31;
  _Float16* pkInp = (_Float16*)pb;                          // 4 * 288 * 512 halves
  _Float16* pkQkv = pkInp + (size_t)4 * NFRAG * FRAGSZ;     // 12 * 288 * 512 halves

  // 1) squared norms + input fragment packing for the cost GEMM
  sqnorm_kernel<<<(2 * B * N + 255) / 256, 256, 0, stream>>>(inp_a, inp_b, sqA, sqB);
  pack_inp_kernel<<<dim3(NFRAG, 4), 32, 0, stream>>>(inp_a, inp_b, pkInp);
  // 2) cost matrices into padded K buffers (WMMA on packed f16 fragments)
  cost_kernel<<<dim3(N / 16, N / 16, 4), 32, 0, stream>>>(pkInp, sqA, sqB, Ka, Kb);
  // 3) exact 0.75-quantile of batch-0 cost per direction: radix select, two order stats
  qinit_kernel<<<1, 4, 0, stream>>>(qstate);
  const size_t M = (size_t)N * N;
  for (int slot = 0; slot < 4; ++slot) {
    const float* Km = (slot < 2) ? Ka : Kb; // batch 0 base
    for (int p = 0; p < 4; ++p) {
      unsigned shift = 24u - 8u * p;
      unsigned maskHigh = (p == 0) ? 0u : (0xFFFFFFFFu << (shift + 8u));
      qzero_kernel<<<1, 256, 0, stream>>>(hist);
      qhist_kernel<<<(unsigned)((M + 255) / 256), 256, 0, stream>>>(
          Km, qstate, hist, maskHigh, shift, slot);
      qscan_kernel<<<1, 1, 0, stream>>>(qstate, hist, shift, slot);
    }
  }
  // 4) fill padded borders with vc, corner 0; then K = max(exp(-pc/reg), 1e-8)
  border_kernel<<<(4 * N1 + 255) / 256, 256, 0, stream>>>(Ka, Kb, qstate);
  {
    size_t tot = 2 * SK;
    ktransform_kernel<<<(unsigned)((tot + 255) / 256), 256, 0, stream>>>(Ka);
  }
  // 5) Sinkhorn: 50 iterations, 4 problems (dir x batch) batched per launch.
  initv_kernel<<<(4 * N1 + 255) / 256, 256, 0, stream>>>(v);
  for (int it = 0; it < 50; ++it) {
    ustep_kernel<<<dim3(N1, 4), 256, 0, stream>>>(Ka, Kb, v, u);
    vstep_kernel<<<dim3((N1 + 63) / 64, 4), 256, 0, stream>>>(Ka, Kb, u, v);
  }
  // 6) plan-normalization denominators (v cancels algebraically)
  den_kernel<<<dim3(N / 64, 4), 256, 0, stream>>>(Ka, Kb, u, invden);
  // 7) qkv projections: 1x1 conv then depthwise 3x3, then fragment packing
  const int QT = B * 3 * C * N;
  conv1x1_qkv_kernel<<<(QT + 255) / 256, 256, 0, stream>>>(inp_a, qkvA_w1, tmp1A);
  conv1x1_qkv_kernel<<<(QT + 255) / 256, 256, 0, stream>>>(inp_b, qkvB_w1, tmp1B);
  dwconv_kernel<<<(QT + 255) / 256, 256, 0, stream>>>(tmp1A, qkvA_dw, qkvA);
  dwconv_kernel<<<(QT + 255) / 256, 256, 0, stream>>>(tmp1B, qkvB_dw, qkvB);
  pack_qkv_kernel<<<dim3(NFRAG, 12), 32, 0, stream>>>(qkvA, qkvB, pkQkv);
  // 8) fused flash attention + OT bias (WMMA), both directions & batches
  attn_kernel<<<dim3(N / 16, 4), 32, 0, stream>>>(pkQkv, Ka, Kb, u, invden,
                                                  alpha_p, beta_p, oA, oB);
  // 9) output 1x1 conv + bias + residual
  const int OT = B * C * N;
  outconv_kernel<<<(OT + 255) / 256, 256, 0, stream>>>(oA, outA_w, outA_b, inp_a, out);
  outconv_kernel<<<(OT + 255) / 256, 256, 0, stream>>>(oB, outB_w, outB_b, inp_b,
                                                       out + (size_t)B * C * N);
}
